// Attention_67405216743899
// MI455X (gfx1250) — matrix operand order repaired and checked by reading, not compiled, not run on hardware
//
#include <hip/hip_runtime.h>
#include <hip/hip_bf16.h>

#define NB     2
#define NH     16
#define NQ     256
#define HDIM   64
#define DIMC   1024
#define LPAST  8192
#define LTOT   (LPAST + NQ)      // 8448
#define NPART  4
#define KSPAN  (LTOT / NPART)    // 2112, multiple of 32
#define NTASK  512               // B*H*(NQ/16) wave tasks
#define QK_SCALE 0.125f          // 64^-0.5

typedef __attribute__((ext_vector_type(16))) _Float16 v16h;
typedef __attribute__((ext_vector_type(8)))  float    v8f;
typedef __attribute__((ext_vector_type(4)))  unsigned tdm_g0_t;
typedef __attribute__((ext_vector_type(8)))  int      tdm_g1_t;
typedef __attribute__((ext_vector_type(4)))  int      tdm_g2_t;

static __device__ __forceinline__ v8f wmma16x16x32(v16h a, v16h b, v8f c) {
  return __builtin_amdgcn_wmma_f32_16x16x32_f16(false, a, false, b, (short)0, c, false, false);
}

static __device__ __forceinline__ unsigned pack2(_Float16 lo, _Float16 hi) {
  return (unsigned)__builtin_bit_cast(unsigned short, lo) |
         ((unsigned)__builtin_bit_cast(unsigned short, hi) << 16);
}

static __device__ __forceinline__ unsigned lds_off(const void* p) {
  return (unsigned)(uintptr_t)p;   // low 32 bits of a generic LDS pointer = LDS byte address
}

// TDM: 2D f16 tile (tile_d0 halves fast dim x tile_d1 rows) global -> LDS.
// LDS padding: +16B after every 64B row segment -> matches LDSTRIDE=40 (32+8 halves).
static __device__ __forceinline__ void tdm_load_tile_f16(unsigned lds_addr,
                                                         const _Float16* gsrc,
                                                         unsigned tensor_d0, unsigned tensor_d1,
                                                         unsigned tile_d0, unsigned tile_d1,
                                                         unsigned long long d0_stride) {
  const unsigned long long ga = (unsigned long long)(uintptr_t)gsrc;
  tdm_g0_t g0;
  g0[0] = 1u;                                             // count=1, user descriptor
  g0[1] = lds_addr;                                       // LDS byte address
  g0[2] = (unsigned)(ga & 0xFFFFFFFFu);                   // global_addr[31:0]
  g0[3] = (unsigned)((ga >> 32) & 0x01FFFFFFu) | (2u << 30);  // [56:32] | type=2
  tdm_g1_t g1;
  // data_size=1 (2B), pad_enable, pad_interval=3 (64B), pad_amount=3 (4 DWORDs)
  g1[0] = (int)((1u << 16) | (1u << 20) | (3u << 22) | (3u << 25));
  g1[1] = (int)((tensor_d0 & 0xFFFFu) << 16);             // tensor_dim0[15:0] @bit48
  g1[2] = (int)(((tensor_d0 >> 16) & 0xFFFFu) | ((tensor_d1 & 0xFFFFu) << 16));
  g1[3] = (int)(((tensor_d1 >> 16) & 0xFFFFu) | ((tile_d0 & 0xFFFFu) << 16));
  g1[4] = (int)(tile_d1 & 0xFFFFu);                       // tile_dim1; tile_dim2=0
  g1[5] = (int)(unsigned)(d0_stride & 0xFFFFFFFFull);     // tensor_dim0_stride[31:0]
  g1[6] = (int)(unsigned)((d0_stride >> 32) & 0xFFFFull); // [47:32]; dim1_stride=0
  g1[7] = 0;
  const tdm_g2_t z4 = {0, 0, 0, 0};
  const tdm_g1_t z8 = {0, 0, 0, 0, 0, 0, 0, 0};
  // 6-arg toolchain variant: (g0, g1, g2, g3, g4, cpol)
  __builtin_amdgcn_tensor_load_to_lds(g0, g1, z4, z4, z8, 0);
}

// A fragment: 16(M) x 32(K); per-lane halves [0..7]@k=g*8, [8..15]@k=16+g*8 -> 2x b128.
static __device__ __forceinline__ v16h load_af(const _Float16* src, int stride) {
  const int lane = threadIdx.x & 31;
  const _Float16* p = src + (lane & 15) * stride + (lane >> 4) * 8;
  v16h a;
#pragma unroll
  for (int i = 0; i < 8; ++i) a[i] = p[i];
#pragma unroll
  for (int i = 0; i < 8; ++i) a[8 + i] = p[16 + i];
  return a;
}

// B fragment: 32(K) x 16(N) from TRANSPOSED storage (k contiguous per lane) -> 2x b128.
static __device__ __forceinline__ v16h load_bt(const _Float16* src, int stride) {
  const int lane = threadIdx.x & 31;
  const _Float16* p = src + (lane & 15) * stride + (lane >> 4) * 8;
  v16h b;
#pragma unroll
  for (int i = 0; i < 8; ++i) b[i] = p[i];
#pragma unroll
  for (int i = 0; i < 8; ++i) b[8 + i] = p[16 + i];
  return b;
}

__global__ void cvt_f32_f16(const float* __restrict__ src, _Float16* __restrict__ dst, int n) {
  for (int i = blockIdx.x * blockDim.x + threadIdx.x; i < n; i += gridDim.x * blockDim.x)
    dst[i] = (_Float16)src[i];
}

// wT[n][k] (f16) = w[k][n] (f32); tiled so both sides stay coalesced.
__global__ __launch_bounds__(256) void cvt_transpose(const float* __restrict__ w,
                                                     _Float16* __restrict__ wT,
                                                     int K, int N) {
  __shared__ _Float16 tile[32][33];
  const int k0 = blockIdx.y * 32, n0 = blockIdx.x * 32;
  const int tr = threadIdx.x >> 5, tc = threadIdx.x & 31;
#pragma unroll
  for (int q = 0; q < 4; ++q)
    tile[tr + 8 * q][tc] = (_Float16)w[(size_t)(k0 + tr + 8 * q) * N + n0 + tc];
  __syncthreads();
#pragma unroll
  for (int q = 0; q < 4; ++q)
    wT[(size_t)(n0 + tr + 8 * q) * K + k0 + tc] = tile[tc][tr + 8 * q];
}

#define LDSTRIDE 40   // 32 + 8 halves: bank spread, 16B-aligned b128 segments

// ---------------- QKV projection GEMM: TDM double-buffered tiles ----------------
__global__ __launch_bounds__(128) void qkv_gemm(const _Float16* __restrict__ A,
                                                const _Float16* __restrict__ BT, // [3072][1024]
                                                const float* __restrict__ bias,
                                                _Float16* __restrict__ qf,
                                                _Float16* __restrict__ kf,
                                                _Float16* __restrict__ vf) {
  __shared__ alignas(16) _Float16 As[2][64 * LDSTRIDE];
  __shared__ alignas(16) _Float16 Bt[2][64 * LDSTRIDE];
  const int n0 = blockIdx.x * 64;
  const int m0 = blockIdx.y * 64;
  const int t = threadIdx.x, w = t >> 5, lane = t & 31;
  const int g = lane >> 4, ln = lane & 15;

  v8f acc[4];
#pragma unroll
  for (int j = 0; j < 4; ++j) acc[j] = {};

  if (w == 0) {   // wave 0 drives the Tensor Data Mover
    tdm_load_tile_f16(lds_off(&As[0][0]), A + (size_t)m0 * DIMC, DIMC, 512, 32, 64, DIMC);
    tdm_load_tile_f16(lds_off(&Bt[0][0]), BT + (size_t)n0 * DIMC, DIMC, 3072, 32, 64, DIMC);
    __builtin_amdgcn_s_wait_tensorcnt(0);
  }
  __syncthreads();

  int cur = 0;
  for (int kk = 0; kk < DIMC; kk += 32) {
    if (kk + 32 < DIMC && w == 0) {   // prefetch next K-slab into the other buffer
      tdm_load_tile_f16(lds_off(&As[cur ^ 1][0]), A + (size_t)m0 * DIMC + kk + 32,
                        DIMC, 512, 32, 64, DIMC);
      tdm_load_tile_f16(lds_off(&Bt[cur ^ 1][0]), BT + (size_t)n0 * DIMC + kk + 32,
                        DIMC, 3072, 32, 64, DIMC);
    }
    const v16h a  = load_af(&As[cur][(w * 16) * LDSTRIDE], LDSTRIDE);
    const v16h b0 = load_bt(&Bt[cur][0], LDSTRIDE);
    const v16h b1 = load_bt(&Bt[cur][16 * LDSTRIDE], LDSTRIDE);
    const v16h b2 = load_bt(&Bt[cur][32 * LDSTRIDE], LDSTRIDE);
    const v16h b3 = load_bt(&Bt[cur][48 * LDSTRIDE], LDSTRIDE);
    acc[0] = wmma16x16x32(a, b0, acc[0]);
    acc[1] = wmma16x16x32(a, b1, acc[1]);
    acc[2] = wmma16x16x32(a, b2, acc[2]);
    acc[3] = wmma16x16x32(a, b3, acc[3]);
    if (w == 0) __builtin_amdgcn_s_wait_tensorcnt(0);
    __syncthreads();
    cur ^= 1;
  }

#pragma unroll
  for (int j = 0; j < 4; ++j) {
    const int cbase = n0 + j * 16;          // 16-col tile never straddles 64/1024 bounds
    const int tsel = cbase >> 10;
    const int h = (cbase & 1023) >> 6;
    const int c = cbase + ln;
    const float bv = bias[c];
    _Float16* dst = (tsel == 0) ? qf : ((tsel == 1) ? kf : vf);
    const float scale = (tsel == 0) ? QK_SCALE : 1.0f;
    const int dcol = c & 63;
#pragma unroll
    for (int v = 0; v < 8; ++v) {
      const int row = m0 + w * 16 + v + 8 * g;
      const int bb = row >> 8, nq = row & 255;
      dst[((size_t)((bb * NH + h) * NQ + nq)) * HDIM + dcol] =
          (_Float16)((acc[j][v] + bv) * scale);
    }
  }
}

// ---------------- Flash attention, split-KV ----------------
#define KTSTRIDE 72
__global__ __launch_bounds__(128) void flash_attn(const float* __restrict__ pk,
                                                  const float* __restrict__ pv,
                                                  const _Float16* __restrict__ qf,
                                                  const _Float16* __restrict__ kf,
                                                  const _Float16* __restrict__ vf,
                                                  float* __restrict__ Op,
                                                  float* __restrict__ Mp,
                                                  float* __restrict__ Lp) {
  __shared__ alignas(16) _Float16 Kt[32 * KTSTRIDE];   // [key][d]
  __shared__ alignas(16) _Float16 Vt[64 * LDSTRIDE];   // [d][key]
  __shared__ alignas(16) _Float16 Pb[4][16 * LDSTRIDE];
  const int blk = blockIdx.x;
  const int part = blockIdx.y;
  const int qblk = blk & 3;
  const int h = (blk >> 2) & 15;
  const int b = blk >> 6;
  const int t = threadIdx.x, w = t >> 5, lane = t & 31;
  const int g = lane >> 4, ln = lane & 15;
  const int nq0 = qblk * 64 + w * 16;
  const size_t bh = (size_t)(b * NH + h);
  const int task = blk * 4 + w;

  const _Float16* qp = qf + (bh * NQ + nq0) * HDIM;
  const v16h qa0 = load_af(qp, HDIM);
  const v16h qa1 = load_af(qp + 32, HDIM);

  v8f o[4];
#pragma unroll
  for (int j = 0; j < 4; ++j) o[j] = {};
  float ms[8], ls[8];
#pragma unroll
  for (int v = 0; v < 8; ++v) { ms[v] = -1e30f; ls[v] = 0.0f; }

  const int krow = t >> 2, kcol = (t & 3) * 16;   // K loader: one key, 16 d
  const int vd = t & 63, vg = t >> 6;             // V loader: one d, 16 keys

  const int l_end = part * KSPAN + KSPAN;
  for (int l0 = part * KSPAN; l0 < l_end; l0 += 32) {
    if (l0 < LPAST) {           // fp32 past cache: convert+pack while staging
      const float4* k4 =
          reinterpret_cast<const float4*>(pk + (bh * LPAST + (l0 + krow)) * HDIM + kcol);
      float4 f0 = k4[0], f1 = k4[1], f2 = k4[2], f3 = k4[3];
      uint4 q0 = {pack2((_Float16)f0.x, (_Float16)f0.y), pack2((_Float16)f0.z, (_Float16)f0.w),
                  pack2((_Float16)f1.x, (_Float16)f1.y), pack2((_Float16)f1.z, (_Float16)f1.w)};
      uint4 q1 = {pack2((_Float16)f2.x, (_Float16)f2.y), pack2((_Float16)f2.z, (_Float16)f2.w),
                  pack2((_Float16)f3.x, (_Float16)f3.y), pack2((_Float16)f3.z, (_Float16)f3.w)};
      uint4* kd = reinterpret_cast<uint4*>(&Kt[krow * KTSTRIDE + kcol]);
      kd[0] = q0; kd[1] = q1;
      const float* vp = pv + (bh * LPAST + (size_t)(l0 + vg * 16)) * HDIM + vd;
      unsigned u[8];
#pragma unroll
      for (int i = 0; i < 8; ++i)
        u[i] = pack2((_Float16)vp[(2 * i) * HDIM], (_Float16)vp[(2 * i + 1) * HDIM]);
      uint4* vdst = reinterpret_cast<uint4*>(&Vt[vd * LDSTRIDE + vg * 16]);
      vdst[0] = uint4{u[0], u[1], u[2], u[3]};
      vdst[1] = uint4{u[4], u[5], u[6], u[7]};
    } else {                    // new keys already f16
      const uint4* k4 =
          reinterpret_cast<const uint4*>(kf + (bh * NQ + (l0 - LPAST + krow)) * HDIM + kcol);
      uint4* kd = reinterpret_cast<uint4*>(&Kt[krow * KTSTRIDE + kcol]);
      kd[0] = k4[0]; kd[1] = k4[1];
      const _Float16* vp = vf + (bh * NQ + (size_t)(l0 - LPAST + vg * 16)) * HDIM + vd;
      unsigned u[8];
#pragma unroll
      for (int i = 0; i < 8; ++i)
        u[i] = pack2(vp[(2 * i) * HDIM], vp[(2 * i + 1) * HDIM]);
      uint4* vdst = reinterpret_cast<uint4*>(&Vt[vd * LDSTRIDE + vg * 16]);
      vdst[0] = uint4{u[0], u[1], u[2], u[3]};
      vdst[1] = uint4{u[4], u[5], u[6], u[7]};
    }
    if (l0 + 32 < l_end && l0 + 32 < LPAST) {   // pull next chunk toward L2
      __builtin_prefetch(pk + (bh * LPAST + (l0 + 32 + krow)) * HDIM + kcol, 0, 1);
      __builtin_prefetch(pv + (bh * LPAST + (l0 + 32 + vg * 16)) * HDIM + vd, 0, 1);
    }
    __syncthreads();

    // S = Q K^T : load all fragments, then issue the 4 wmma
    const v16h kb0 = load_bt(&Kt[0], KTSTRIDE);
    const v16h kb1 = load_bt(&Kt[32], KTSTRIDE);
    const v16h kb2 = load_bt(&Kt[16 * KTSTRIDE], KTSTRIDE);
    const v16h kb3 = load_bt(&Kt[16 * KTSTRIDE + 32], KTSTRIDE);
    v8f s0 = {}, s1 = {};
    s0 = wmma16x16x32(qa0, kb0, s0);
    s0 = wmma16x16x32(qa1, kb1, s0);
    s1 = wmma16x16x32(qa0, kb2, s1);
    s1 = wmma16x16x32(qa1, kb3, s1);

    float alpha[8];
#pragma unroll
    for (int v = 0; v < 8; ++v) {
      float mx = fmaxf(s0[v], s1[v]);
      mx = fmaxf(mx, __shfl_xor(mx, 1));
      mx = fmaxf(mx, __shfl_xor(mx, 2));
      mx = fmaxf(mx, __shfl_xor(mx, 4));
      mx = fmaxf(mx, __shfl_xor(mx, 8));
      const float mnew = fmaxf(ms[v], mx);
      const float al = __expf(ms[v] - mnew);
      const float p0 = __expf(s0[v] - mnew);
      const float p1 = __expf(s1[v] - mnew);
      float rs = p0 + p1;
      rs += __shfl_xor(rs, 1); rs += __shfl_xor(rs, 2);
      rs += __shfl_xor(rs, 4); rs += __shfl_xor(rs, 8);
      ls[v] = ls[v] * al + rs;
      ms[v] = mnew;
      alpha[v] = al;
      const int prow = v + 8 * g;
      Pb[w][prow * LDSTRIDE + ln]      = (_Float16)p0;
      Pb[w][prow * LDSTRIDE + 16 + ln] = (_Float16)p1;
    }
    asm volatile("s_wait_dscnt 0" ::: "memory");   // same-wave cross-lane LDS RAW
    const v16h pa  = load_af(&Pb[w][0], LDSTRIDE);
    const v16h vb0 = load_bt(&Vt[0], LDSTRIDE);
    const v16h vb1 = load_bt(&Vt[16 * LDSTRIDE], LDSTRIDE);
    const v16h vb2 = load_bt(&Vt[32 * LDSTRIDE], LDSTRIDE);
    const v16h vb3 = load_bt(&Vt[48 * LDSTRIDE], LDSTRIDE);
#pragma unroll
    for (int j = 0; j < 4; ++j)
#pragma unroll
      for (int v = 0; v < 8; ++v) o[j][v] *= alpha[v];
    o[0] = wmma16x16x32(pa, vb0, o[0]);
    o[1] = wmma16x16x32(pa, vb1, o[1]);
    o[2] = wmma16x16x32(pa, vb2, o[2]);
    o[3] = wmma16x16x32(pa, vb3, o[3]);
    __syncthreads();
  }

#pragma unroll
  for (int v = 0; v < 8; ++v) {
    const int row = v + 8 * g;
#pragma unroll
    for (int j = 0; j < 4; ++j)
      Op[(((size_t)(part * NTASK + task)) * 16 + row) * HDIM + j * 16 + ln] = o[j][v];
    if (ln == 0) {
      Mp[(part * NTASK + task) * 16 + row] = ms[v];
      Lp[(part * NTASK + task) * 16 + row] = ls[v];
    }
  }
}

// ---------------- Split-KV combine (log-sum-exp merge) ----------------
__global__ __launch_bounds__(256) void combine(const float* __restrict__ Op,
                                               const float* __restrict__ Mp,
                                               const float* __restrict__ Lp,
                                               _Float16* __restrict__ attnf) {
  const int task = blockIdx.x;
  const int t = threadIdx.x;
  const int r = t >> 4;
  const int d = (t & 15) * 4;
  const int w = task & 3;
  const int blkx = task >> 2;
  const int qblk = blkx & 3, h = (blkx >> 2) & 15, b = blkx >> 6;
  const int nq = qblk * 64 + w * 16 + r;

  float m[NPART], wgt[NPART];
  float mstar = -1e30f;
#pragma unroll
  for (int p = 0; p < NPART; ++p) {
    m[p] = Mp[(p * NTASK + task) * 16 + r];
    mstar = fmaxf(mstar, m[p]);
  }
  float lsum = 0.f;
#pragma unroll
  for (int p = 0; p < NPART; ++p) {
    wgt[p] = __expf(m[p] - mstar);
    lsum += wgt[p] * Lp[(p * NTASK + task) * 16 + r];
  }
  const float inv = 1.0f / lsum;
  float acc[4] = {0.f, 0.f, 0.f, 0.f};
#pragma unroll
  for (int p = 0; p < NPART; ++p) {
    const float4 ov = *reinterpret_cast<const float4*>(
        &Op[(((size_t)(p * NTASK + task)) * 16 + r) * HDIM + d]);
    acc[0] += wgt[p] * ov.x; acc[1] += wgt[p] * ov.y;
    acc[2] += wgt[p] * ov.z; acc[3] += wgt[p] * ov.w;
  }
#pragma unroll
  for (int i = 0; i < 4; ++i)
    attnf[((size_t)(b * NQ + nq)) * DIMC + h * HDIM + d + i] = (_Float16)(acc[i] * inv);
}

// ---------------- Output projection GEMM: TDM double-buffered tiles ----------------
__global__ __launch_bounds__(128) void proj_gemm(const _Float16* __restrict__ A,
                                                 const _Float16* __restrict__ BT, // [1024][1024]
                                                 const float* __restrict__ bias,
                                                 float* __restrict__ out) {
  __shared__ alignas(16) _Float16 As[2][64 * LDSTRIDE];
  __shared__ alignas(16) _Float16 Bt[2][64 * LDSTRIDE];
  const int n0 = blockIdx.x * 64;
  const int m0 = blockIdx.y * 64;
  const int t = threadIdx.x, w = t >> 5, lane = t & 31;
  const int g = lane >> 4, ln = lane & 15;

  v8f acc[4];
#pragma unroll
  for (int j = 0; j < 4; ++j) acc[j] = {};

  if (w == 0) {
    tdm_load_tile_f16(lds_off(&As[0][0]), A + (size_t)m0 * DIMC, DIMC, 512, 32, 64, DIMC);
    tdm_load_tile_f16(lds_off(&Bt[0][0]), BT + (size_t)n0 * DIMC, DIMC, DIMC, 32, 64, DIMC);
    __builtin_amdgcn_s_wait_tensorcnt(0);
  }
  __syncthreads();

  int cur = 0;
  for (int kk = 0; kk < DIMC; kk += 32) {
    if (kk + 32 < DIMC && w == 0) {
      tdm_load_tile_f16(lds_off(&As[cur ^ 1][0]), A + (size_t)m0 * DIMC + kk + 32,
                        DIMC, 512, 32, 64, DIMC);
      tdm_load_tile_f16(lds_off(&Bt[cur ^ 1][0]), BT + (size_t)n0 * DIMC + kk + 32,
                        DIMC, DIMC, 32, 64, DIMC);
    }
    const v16h a  = load_af(&As[cur][(w * 16) * LDSTRIDE], LDSTRIDE);
    const v16h b0 = load_bt(&Bt[cur][0], LDSTRIDE);
    const v16h b1 = load_bt(&Bt[cur][16 * LDSTRIDE], LDSTRIDE);
    const v16h b2 = load_bt(&Bt[cur][32 * LDSTRIDE], LDSTRIDE);
    const v16h b3 = load_bt(&Bt[cur][48 * LDSTRIDE], LDSTRIDE);
    acc[0] = wmma16x16x32(a, b0, acc[0]);
    acc[1] = wmma16x16x32(a, b1, acc[1]);
    acc[2] = wmma16x16x32(a, b2, acc[2]);
    acc[3] = wmma16x16x32(a, b3, acc[3]);
    if (w == 0) __builtin_amdgcn_s_wait_tensorcnt(0);
    __syncthreads();
    cur ^= 1;
  }

#pragma unroll
  for (int j = 0; j < 4; ++j) {
    const int c = n0 + j * 16 + ln;
    const float bv = bias[c];
#pragma unroll
    for (int v = 0; v < 8; ++v) {
      const int row = m0 + w * 16 + v + 8 * g;
      out[(size_t)row * DIMC + c] = acc[j][v] + bv;
    }
  }
}

extern "C" void kernel_launch(void* const* d_in, const int* in_sizes, int n_in,
                              void* d_out, int out_size, void* d_ws, size_t ws_size,
                              hipStream_t stream) {
  const float* x      = (const float*)d_in[0];
  const float* past_k = (const float*)d_in[1];
  const float* past_v = (const float*)d_in[2];
  const float* w_qkv  = (const float*)d_in[3];
  const float* b_qkv  = (const float*)d_in[4];
  const float* w_proj = (const float*)d_in[5];
  const float* b_proj = (const float*)d_in[6];
  float* out = (float*)d_out;

  _Float16* ws     = (_Float16*)d_ws;
  _Float16* xf     = ws;                          // 512*1024
  _Float16* wqkvT  = xf + 512 * 1024;             // [3072][1024]
  _Float16* wprojT = wqkvT + 3072 * 1024;         // [1024][1024]
  _Float16* qf     = wprojT + 1024 * 1024;
  _Float16* kf     = qf + NB * NH * NQ * HDIM;
  _Float16* vf     = kf + NB * NH * NQ * HDIM;
  _Float16* attnf  = vf + NB * NH * NQ * HDIM;    // 512*1024
  float*    Op     = (float*)(attnf + 512 * 1024);  // [NPART][NTASK][16][64]
  float*    Mp     = Op + NPART * NTASK * 16 * HDIM;
  float*    Lp     = Mp + NPART * NTASK * 16;

  cvt_f32_f16<<<512, 256, 0, stream>>>(x, xf, 512 * 1024);
  cvt_transpose<<<dim3(3072 / 32, 1024 / 32), 256, 0, stream>>>(w_qkv, wqkvT, DIMC, 3072);
  cvt_transpose<<<dim3(1024 / 32, 1024 / 32), 256, 0, stream>>>(w_proj, wprojT, DIMC, DIMC);

  qkv_gemm<<<dim3(3072 / 64, 512 / 64), 128, 0, stream>>>(xf, wqkvT, b_qkv, qf, kf, vf);
  flash_attn<<<dim3(128, NPART), 128, 0, stream>>>(past_k, past_v, qf, kf, vf, Op, Mp, Lp);
  combine<<<NTASK, 256, 0, stream>>>(Op, Mp, Lp, attnf);
  proj_gemm<<<dim3(1024 / 64, 512 / 64), 128, 0, stream>>>(attnf, wprojT, b_proj, out);
}
